// Attention_8796093022669
// MI455X (gfx1250) — matrix operand order repaired and checked by reading, not compiled, not run on hardware
//
#include <hip/hip_runtime.h>

// ---------- types ----------
typedef __attribute__((ext_vector_type(16))) __bf16        v16bf;
typedef __attribute__((ext_vector_type(8)))  float         v8f;
typedef __attribute__((ext_vector_type(4)))  unsigned int  u32x4;
typedef __attribute__((ext_vector_type(4)))  float         f32x4;
typedef __attribute__((ext_vector_type(4)))  unsigned short u16x4;
typedef int v4i __attribute__((vector_size(16)));   // matches clang builtin "V4i"

union Frag { v16bf v; u32x4 u[2]; };

// ---------- problem constants ----------
#define BSZ   2
#define SEQ   2048
#define DIM   4096
#define NH    32
#define NKV   8
#define HD    128
#define NREP  4

__device__ __forceinline__ unsigned short f2bf(float f) {
  unsigned u = __builtin_bit_cast(unsigned, f);
  unsigned r = u + 0x7FFFu + ((u >> 16) & 1u);   // round-to-nearest-even
  return (unsigned short)(r >> 16);
}

// ---------- async memory->LDS b128 copy (CDNA5), with portable fallback ----
#if __has_builtin(__builtin_amdgcn_global_load_async_to_lds_b128)
#define HAVE_ASYNC_LDS 1
__device__ __forceinline__ void async_copy_b128(const unsigned short* g, unsigned short* l) {
  __builtin_amdgcn_global_load_async_to_lds_b128(
      (__attribute__((address_space(1))) v4i*)(g),
      (__attribute__((address_space(3))) v4i*)(l), 0, 0);
}
__device__ __forceinline__ void async_wait0() {
#if __has_builtin(__builtin_amdgcn_s_wait_asynccnt)
  __builtin_amdgcn_s_wait_asynccnt(0);
#else
  asm volatile("s_wait_asynccnt 0" ::: "memory");
#endif
}
#else
#define HAVE_ASYNC_LDS 0
__device__ __forceinline__ void async_copy_b128(const unsigned short* g, unsigned short* l) {
  *(u32x4*)l = *(const u32x4*)g;
}
__device__ __forceinline__ void async_wait0() {}
#endif

// =====================================================================
// 1) fp32 -> bf16 cast (vectorized)
// =====================================================================
__global__ __launch_bounds__(256) void cast_f32_bf16(
    const float* __restrict__ in, unsigned short* __restrict__ out, int n4) {
  int i = blockIdx.x * 256 + threadIdx.x;
  if (i >= n4) return;
  f32x4 f = ((const f32x4*)in)[i];
  u16x4 r;
  r.x = f2bf(f.x); r.y = f2bf(f.y); r.z = f2bf(f.z); r.w = f2bf(f.w);
  ((u16x4*)out)[i] = r;
}

// =====================================================================
// 2) bf16 GEMM:  C(f32, MxN) = A(bf16, MxK, row-major) * B(bf16, KxN, row-major)
//    block tile 128x128, K-step 32, 8 waves, wave tile 32x64 (2x4 wmma tiles)
//    N, K compile-time -> immediate-offset epilogue stores
// =====================================================================
#define BM 128
#define BN 128
#define BK 32
#define LDT 40          // padded k-stride in shorts (80B, 16B multiple)

template <int N, int K>
__global__ __launch_bounds__(256) void gemm_bf16(
    const unsigned short* __restrict__ A, const unsigned short* __restrict__ B,
    float* __restrict__ C) {
  __shared__ unsigned short As[BM * LDT];
  __shared__ unsigned short Bs[BN * LDT];

  const int tid  = threadIdx.x;
  const int lane = tid & 31;
  const int wave = tid >> 5;
  const int lmod = lane & 15, lhalf = lane >> 4;
  const int wm = (wave & 3) * 32, wn = (wave >> 2) * 64;
  const int bm = blockIdx.y * BM, bn = blockIdx.x * BN;

  v8f zer = {0.f,0.f,0.f,0.f,0.f,0.f,0.f,0.f};
  v8f acc[2][4];
#pragma unroll
  for (int i = 0; i < 2; ++i)
#pragma unroll
    for (int j = 0; j < 4; ++j) acc[i][j] = zer;

  for (int k0 = 0; k0 < K; k0 += BK) {
    __syncthreads();
    // A tile 128x32 -> LDS [m][k] : straight copy, async path
#pragma unroll
    for (int it = 0; it < 2; ++it) {
      int idx = tid + it * 256;             // 0..511
      int row = idx >> 2, seg = idx & 3;
      async_copy_b128(A + (size_t)(bm + row) * K + k0 + seg * 8,
                      As + row * LDT + seg * 8);
    }
    // B tile 32x128 -> LDS transposed [n][k] (must flow through VGPRs)
#pragma unroll
    for (int it = 0; it < 2; ++it) {
      int idx = tid + it * 256;             // 0..511
      int kr = idx >> 4, seg = idx & 15;
      u32x4 d = *(const u32x4*)(B + (size_t)(k0 + kr) * N + bn + seg * 8);
      const unsigned short* p = (const unsigned short*)&d;
#pragma unroll
      for (int j = 0; j < 8; ++j) Bs[(seg * 8 + j) * LDT + kr] = p[j];
    }
    async_wait0();
    __syncthreads();

    Frag a[2], b[4];
#pragma unroll
    for (int mt = 0; mt < 2; ++mt) {
      int m = wm + mt * 16 + lmod;
      a[mt].u[0] = *(const u32x4*)(As + m * LDT + lhalf * 8);        // k 0..7 (+8*lhalf)
      a[mt].u[1] = *(const u32x4*)(As + m * LDT + lhalf * 8 + 16);   // k 16..23 (+8*lhalf)
    }
#pragma unroll
    for (int nt = 0; nt < 4; ++nt) {
      int n = wn + nt * 16 + lmod;
      b[nt].u[0] = *(const u32x4*)(Bs + n * LDT + lhalf * 8);        // k 0..7 (+8*lhalf)
      b[nt].u[1] = *(const u32x4*)(Bs + n * LDT + lhalf * 8 + 16);   // k 16..23 (+8*lhalf)
    }
#pragma unroll
    for (int mt = 0; mt < 2; ++mt)
#pragma unroll
      for (int nt = 0; nt < 4; ++nt)
        acc[mt][nt] = __builtin_amdgcn_wmma_f32_16x16x32_bf16(
            false, a[mt].v, false, b[nt].v, (short)0, acc[mt][nt], false, false);
  }

  // epilogue: C layout M = v + 8*lhalf, N = lmod; one base + immediate offsets
#pragma unroll
  for (int mt = 0; mt < 2; ++mt)
#pragma unroll
    for (int nt = 0; nt < 4; ++nt) {
      int row0 = bm + wm + mt * 16 + 8 * lhalf;
      int col  = bn + wn + nt * 16 + lmod;
      float* cp = C + (size_t)row0 * N + col;
#pragma unroll
      for (int v = 0; v < 8; ++v) cp[(size_t)v * N] = acc[mt][nt][v];
    }
}

// =====================================================================
// 3) RoPE: src fp32 [b, s, nh, 128] -> dst bf16 [b, nh, s, 128]
// =====================================================================
__global__ __launch_bounds__(256) void rope_kernel(
    const float* __restrict__ src, unsigned short* __restrict__ dst,
    const float* __restrict__ fcos, const float* __restrict__ fsin, int nheads) {
  int idx = blockIdx.x * 256 + threadIdx.x;        // over B*S*nheads*64
  int i = idx & 63;
  int t = idx >> 6;
  int h = t % nheads; t /= nheads;
  int s = t % SEQ;
  int b = t / SEQ;
  size_t row = (size_t)(b * SEQ + s) * ((size_t)nheads * HD);
  float re = src[row + h * HD + 2 * i];
  float im = src[row + h * HD + 2 * i + 1];
  float c  = fcos[s * 64 + i];
  float sn = fsin[s * 64 + i];
  float ore = re * c - im * sn;
  float oim = re * sn + im * c;
  size_t o = ((size_t)(b * nheads + h) * SEQ + s) * HD + 2 * i;
  dst[o]     = f2bf(ore);
  dst[o + 1] = f2bf(oim);
}

// =====================================================================
// 4) V cast+transpose: fp32 [b, s, kvh, 128] -> bf16 [b, kvh, s, 128]
// =====================================================================
__global__ __launch_bounds__(256) void vcast_kernel(
    const float* __restrict__ src, unsigned short* __restrict__ dst) {
  int idx = blockIdx.x * 256 + threadIdx.x;        // over B*S*NKV*128
  int d = idx & 127;
  int t = idx >> 7;
  int h = t % NKV; t /= NKV;
  int s = t % SEQ;
  int b = t / SEQ;
  float v = src[((size_t)(b * SEQ + s) * NKV + h) * HD + d];
  dst[((size_t)(b * NKV + h) * SEQ + s) * HD + d] = f2bf(v);
}

// =====================================================================
// 5) Flash attention (causal, GQA). One block = (b, head, 128 q-rows).
//    8 waves x 16 q-rows. Key blocks of 64. bf16 wmma, fp32 accum.
// =====================================================================
#define FKB 64
#define KLD 136     // [key][d]  d-stride (272B, 16B mult)
#define VLD 72      // [d][key]  key-stride (144B, 16B mult)
#define PLD 72      // per-wave P scratch [m][key]

__global__ __launch_bounds__(256) void flash_kernel(
    const unsigned short* __restrict__ Q,   // [b, h,  s, 128]
    const unsigned short* __restrict__ Kc,  // [b, kvh, s, 128]
    const unsigned short* __restrict__ Vc,  // [b, kvh, s, 128]
    unsigned short* __restrict__ O) {       // [b, s, h*128+d]
  __shared__ unsigned short Ks[FKB * KLD];
  __shared__ unsigned short Vs[HD * VLD];
  __shared__ unsigned short Ps[8 * 16 * PLD];

  const int tid  = threadIdx.x;
  const int lane = tid & 31;
  const int wave = tid >> 5;
  const int lmod = lane & 15, lhalf = lane >> 4;
  const int qtile = blockIdx.x;
  const int h = blockIdx.y;
  const int b = blockIdx.z;
  const int g = h >> 2;                      // kv head (NREP=4)
  const int qbase = qtile * 128;
  const int qrow_w = qbase + wave * 16;
  const float scale = 0.08838834764831845f;  // 1/sqrt(128)

  // load Q fragments (held for whole kernel): A-layout, K-dim = head_dim
  Frag qa[4];
  const unsigned short* qp = Q + (size_t)(b * NH + h) * SEQ * HD;
#pragma unroll
  for (int kk = 0; kk < 4; ++kk) {
    size_t base = (size_t)(qrow_w + lmod) * HD + kk * 32 + lhalf * 8;
    qa[kk].u[0] = *(const u32x4*)(qp + base);
    qa[kk].u[1] = *(const u32x4*)(qp + base + 16);
  }

  v8f zer = {0.f,0.f,0.f,0.f,0.f,0.f,0.f,0.f};
  v8f o[8];
#pragma unroll
  for (int dt = 0; dt < 8; ++dt) o[dt] = zer;
  float mrow[8], lrow[8];
#pragma unroll
  for (int v = 0; v < 8; ++v) { mrow[v] = -1e30f; lrow[v] = 0.f; }

  const unsigned short* kp = Kc + (size_t)(b * NKV + g) * SEQ * HD;
  const unsigned short* vp = Vc + (size_t)(b * NKV + g) * SEQ * HD;

  const int nblk = 2 * (qtile + 1);          // (qbase+128)/64 causal key blocks
  for (int kb = 0; kb < nblk; ++kb) {
    const int kbase = kb * FKB;
    __syncthreads();
    // K block 64x128 -> LDS [key][d] : straight copy, async path
#pragma unroll
    for (int it = 0; it < 4; ++it) {
      int idx = tid + it * 256;              // 0..1023
      int key = idx >> 4, seg = idx & 15;
      async_copy_b128(kp + (size_t)(kbase + key) * HD + seg * 8,
                      Ks + key * KLD + seg * 8);
    }
    // V block 64x128 -> LDS transposed [d][key] (through VGPRs)
#pragma unroll
    for (int it = 0; it < 4; ++it) {
      int idx = tid + it * 256;
      int key = idx >> 4, seg = idx & 15;
      u32x4 dv = *(const u32x4*)(vp + (size_t)(kbase + key) * HD + seg * 8);
      const unsigned short* p = (const unsigned short*)&dv;
#pragma unroll
      for (int j = 0; j < 8; ++j) Vs[(seg * 8 + j) * VLD + key] = p[j];
    }
    async_wait0();
    __syncthreads();

    // S = (Q Kt) * scale, 16 x 64 per wave, 4 n-tiles
    v8f st[4];
#pragma unroll
    for (int nt = 0; nt < 4; ++nt) {
      v8f s = zer;
      int n = nt * 16 + lmod;                // key column held by this lane
#pragma unroll
      for (int kk = 0; kk < 4; ++kk) {
        Frag bk;
        int dof = kk * 32 + lhalf * 8;
        bk.u[0] = *(const u32x4*)(Ks + n * KLD + dof);
        bk.u[1] = *(const u32x4*)(Ks + n * KLD + dof + 16);
        s = __builtin_amdgcn_wmma_f32_16x16x32_bf16(
            false, qa[kk].v, false, bk.v, (short)0, s, false, false);
      }
      st[nt] = s;
    }

    // scale + causal mask + block row-max
    const bool need_mask = (kbase + FKB - 1) > qbase;
    float mblk[8];
#pragma unroll
    for (int v = 0; v < 8; ++v) mblk[v] = -1e30f;
#pragma unroll
    for (int nt = 0; nt < 4; ++nt) {
      int kcol = kbase + nt * 16 + lmod;
#pragma unroll
      for (int v = 0; v < 8; ++v) {
        float sv = st[nt][v] * scale;
        int qr = qrow_w + v + 8 * lhalf;
        if (need_mask && (kcol > qr)) sv = -1e30f;
        st[nt][v] = sv;
        mblk[v] = fmaxf(mblk[v], sv);
      }
    }
#pragma unroll
    for (int v = 0; v < 8; ++v) {
      float x = mblk[v];
      x = fmaxf(x, __shfl_xor(x, 1, 16));
      x = fmaxf(x, __shfl_xor(x, 2, 16));
      x = fmaxf(x, __shfl_xor(x, 4, 16));
      x = fmaxf(x, __shfl_xor(x, 8, 16));
      mblk[v] = x;
    }

    // online softmax update + write P (bf16) to per-wave LDS scratch
#pragma unroll
    for (int v = 0; v < 8; ++v) {
      float mnew  = fmaxf(mrow[v], mblk[v]);
      float alpha = __expf(mrow[v] - mnew);
      mrow[v] = mnew;
      lrow[v] *= alpha;
#pragma unroll
      for (int dt = 0; dt < 8; ++dt) o[dt][v] *= alpha;
      float rs = 0.f;
      int m = wave * 16 + v + 8 * lhalf;
#pragma unroll
      for (int nt = 0; nt < 4; ++nt) {
        float p = __expf(st[nt][v] - mnew);
        rs += p;
        Ps[m * PLD + nt * 16 + lmod] = f2bf(p);
      }
      rs += __shfl_xor(rs, 1, 16);
      rs += __shfl_xor(rs, 2, 16);
      rs += __shfl_xor(rs, 4, 16);
      rs += __shfl_xor(rs, 8, 16);
      lrow[v] += rs;
    }
    __builtin_amdgcn_wave_barrier();         // same-wave DS ordering is in-order

    // O += P @ V : P as A-frags (K = 64 keys -> 2 k-steps)
    Frag pa[2];
#pragma unroll
    for (int kk = 0; kk < 2; ++kk) {
      size_t base = (size_t)(wave * 16 + lmod) * PLD + kk * 32 + lhalf * 8;
      pa[kk].u[0] = *(const u32x4*)(Ps + base);
      pa[kk].u[1] = *(const u32x4*)(Ps + base + 16);
    }
#pragma unroll
    for (int dt = 0; dt < 8; ++dt) {
#pragma unroll
      for (int kk = 0; kk < 2; ++kk) {
        Frag bv;
        int kof = kk * 32 + lhalf * 8;
        bv.u[0] = *(const u32x4*)(Vs + (dt * 16 + lmod) * VLD + kof);
        bv.u[1] = *(const u32x4*)(Vs + (dt * 16 + lmod) * VLD + kof + 16);
        o[dt] = __builtin_amdgcn_wmma_f32_16x16x32_bf16(
            false, pa[kk].v, false, bv.v, (short)0, o[dt], false, false);
      }
    }
  }

  // normalize and store bf16 attention output [b, s, h*128+d]
#pragma unroll
  for (int v = 0; v < 8; ++v) {
    float inv = 1.0f / lrow[v];
    int srow = qrow_w + v + 8 * lhalf;
    size_t rowo = ((size_t)b * SEQ + srow) * (NH * HD) + (size_t)h * HD;
#pragma unroll
    for (int dt = 0; dt < 8; ++dt)
      O[rowo + dt * 16 + lmod] = f2bf(o[dt][v] * inv);
  }
}

// =====================================================================
// launch
// =====================================================================
extern "C" void kernel_launch(void* const* d_in, const int* in_sizes, int n_in,
                              void* d_out, int out_size, void* d_ws, size_t ws_size,
                              hipStream_t stream) {
  const float* x    = (const float*)d_in[0];
  // d_in[1] = start_pos (0), d_in[4] = mask (causal, applied analytically)
  const float* fcos = (const float*)d_in[2];
  const float* fsin = (const float*)d_in[3];
  const float* wq   = (const float*)d_in[5];
  const float* wk   = (const float*)d_in[6];
  const float* wv   = (const float*)d_in[7];
  const float* wo   = (const float*)d_in[8];
  float* out = (float*)d_out;

  char* ws = (char*)d_ws;
  size_t off = 0;
  auto alloc = [&](size_t bytes) -> void* {
    void* p = ws + off;
    off += (bytes + 255) & ~(size_t)255;
    return p;
  };
  const size_t M = (size_t)BSZ * SEQ;        // 4096 token rows
  unsigned short* x_bf  = (unsigned short*)alloc(M * DIM * 2);
  unsigned short* wq_bf = (unsigned short*)alloc((size_t)DIM * DIM * 2);
  unsigned short* wk_bf = (unsigned short*)alloc((size_t)DIM * (NKV * HD) * 2);
  unsigned short* wv_bf = (unsigned short*)alloc((size_t)DIM * (NKV * HD) * 2);
  unsigned short* wo_bf = (unsigned short*)alloc((size_t)DIM * DIM * 2);
  unsigned short* q_bf  = (unsigned short*)alloc(M * NH * HD * 2);
  unsigned short* k_bf  = (unsigned short*)alloc(M * NKV * HD * 2);
  unsigned short* v_bf  = (unsigned short*)alloc(M * NKV * HD * 2);
  float* xq_f = (float*)alloc(M * DIM * 4);
  float* xk_f = (float*)alloc(M * (NKV * HD) * 4);
  float* xv_f = (float*)alloc(M * (NKV * HD) * 4);
  unsigned short* attn_bf = (unsigned short*)xq_f;   // alias: xq_f dead after RoPE

  // 1) casts to bf16
  cast_f32_bf16<<<(int)(M * DIM / 4 / 256), 256, 0, stream>>>(x, x_bf, (int)(M * DIM / 4));
  cast_f32_bf16<<<DIM * DIM / 4 / 256, 256, 0, stream>>>(wq, wq_bf, DIM * DIM / 4);
  cast_f32_bf16<<<DIM * NKV * HD / 4 / 256, 256, 0, stream>>>(wk, wk_bf, DIM * NKV * HD / 4);
  cast_f32_bf16<<<DIM * NKV * HD / 4 / 256, 256, 0, stream>>>(wv, wv_bf, DIM * NKV * HD / 4);
  cast_f32_bf16<<<DIM * DIM / 4 / 256, 256, 0, stream>>>(wo, wo_bf, DIM * DIM / 4);

  // 2) QKV projections
  gemm_bf16<DIM, DIM><<<dim3(DIM / BN, (int)(M / BM)), 256, 0, stream>>>(x_bf, wq_bf, xq_f);
  gemm_bf16<NKV * HD, DIM><<<dim3(NKV * HD / BN, (int)(M / BM)), 256, 0, stream>>>(x_bf, wk_bf, xk_f);
  gemm_bf16<NKV * HD, DIM><<<dim3(NKV * HD / BN, (int)(M / BM)), 256, 0, stream>>>(x_bf, wv_bf, xv_f);

  // 3) RoPE (q, k) + V cast, all into [b, head, s, d] bf16
  rope_kernel<<<(int)(M * NH * 64 / 256), 256, 0, stream>>>(xq_f, q_bf, fcos, fsin, NH);
  rope_kernel<<<(int)(M * NKV * 64 / 256), 256, 0, stream>>>(xk_f, k_bf, fcos, fsin, NKV);
  vcast_kernel<<<(int)(M * NKV * HD / 256), 256, 0, stream>>>(xv_f, v_bf);

  // 4) flash attention -> attn_bf [b, s, 4096]
  flash_kernel<<<dim3(SEQ / 128, NH, BSZ), 256, 0, stream>>>(q_bf, k_bf, v_bf, attn_bf);

  // 5) output projection -> d_out fp32
  gemm_bf16<DIM, DIM><<<dim3(DIM / BN, (int)(M / BM)), 256, 0, stream>>>(attn_bf, wo_bf, out);
}
